// MultiHeadAttention_49340584297031
// MI455X (gfx1250) — matrix operand order repaired and checked by reading, not compiled, not run on hardware
//
#include <hip/hip_runtime.h>

// ---------------------------------------------------------------------------
// MHA block for MI455X (gfx1250): bf16 WMMA GEMMs + softmax + layernorm.
// B=2, L=2048, D=1024, H=16, DH=64.  Outputs: [out (B,L,D) f32][attn (32,L,L) f32]
// Round 3: async global->LDS staging with correctly-typed builtin args
// (v4i addrspace(1)* / addrspace(3)*, per hipcc diagnostic from round 2).
// ---------------------------------------------------------------------------

typedef __bf16 bf16_t;
typedef __attribute__((ext_vector_type(16))) __bf16 v16bf;
typedef __attribute__((ext_vector_type(8)))  __bf16 v8bf;
typedef __attribute__((ext_vector_type(8)))  float  v8f;
typedef __attribute__((ext_vector_type(4)))  float  v4f;

constexpr int Bsz  = 2;
constexpr int Lseq = 2048;
constexpr int Dmod = 1024;
constexpr int DH   = 64;
constexpr int Mrows = Bsz * Lseq;            // 4096
constexpr size_t X_ELEMS = (size_t)Mrows * Dmod;      // 4,194,304
constexpr size_t W_ELEMS = (size_t)Dmod * Dmod;       // 1,048,576

// ---------------- CDNA5 async copy (global -> LDS, ASYNCcnt) ----------------

#if defined(__has_builtin)
#  if __has_builtin(__builtin_amdgcn_global_load_async_to_lds_b128)
#    define HAS_ASYNC_LDS 1
#  else
#    define HAS_ASYNC_LDS 0
#  endif
#else
#  define HAS_ASYNC_LDS 0
#endif

typedef int v4i __attribute__((vector_size(4 * sizeof(int))));
typedef __attribute__((address_space(1))) v4i as1_v4i;   // global
typedef __attribute__((address_space(3))) v4i as3_v4i;   // LDS

static __device__ __forceinline__ void async_copy16B(const bf16_t* g, bf16_t* s) {
#if HAS_ASYNC_LDS
    __builtin_amdgcn_global_load_async_to_lds_b128(
        (as1_v4i*)(void*)g, (as3_v4i*)(void*)s, 0, 0);
#else
    *(v8bf*)s = *(const v8bf*)g;
#endif
}

static __device__ __forceinline__ void wait_async0() {
#if HAS_ASYNC_LDS
#  if __has_builtin(__builtin_amdgcn_s_wait_asynccnt)
    __builtin_amdgcn_s_wait_asynccnt(0);
#  else
    asm volatile("s_wait_asynccnt 0x0" ::: "memory");
#  endif
#endif
}

// ---------------- WMMA helpers ---------------------------------------------

static __device__ __forceinline__ v8f wmma_bf16(v16bf a, v16bf b, v8f c) {
    return __builtin_amdgcn_wmma_f32_16x16x32_bf16(
        /*neg_a=*/false, a, /*neg_b=*/false, b,
        /*c_mod=*/(short)0, c, /*reuse_a=*/false, /*reuse_b=*/false);
}

static __device__ __forceinline__ v16bf pack2(v8bf lo, v8bf hi) {
    v16bf f;
#pragma unroll
    for (int i = 0; i < 8; ++i) { f[i] = lo[i]; f[i + 8] = hi[i]; }
    return f;
}

// A-operand (16x32, MxK): lane<16 -> row=lane, K {0..7,16..23}; lane>=16 -> K {8..15,24..31}
static __device__ __forceinline__ v16bf frag_a(const bf16_t* base, int ld, int row0, int lane) {
    int r = lane & 15, hf = lane >> 4;
    const bf16_t* p = base + (size_t)(row0 + r) * ld + hf * 8;
    return pack2(*(const v8bf*)p, *(const v8bf*)(p + 16));
}

// B-operand (32x16, KxN) sourced from a ROW-MAJOR [N,K] matrix (i.e. B = M^T):
// lane holds column n=lane&15; lanes<16 K {0..7,16..23}, lanes>=16 K {8..15,24..31}
static __device__ __forceinline__ v16bf frag_b(const bf16_t* base, int ld, int col0, int lane) {
    int n = lane & 15, hf = lane >> 4;
    const bf16_t* p = base + (size_t)(col0 + n) * ld + hf * 8;
    return pack2(*(const v8bf*)p, *(const v8bf*)(p + 16));
}

// A-operand from fp32 source (attention probs) with on-the-fly bf16 convert
static __device__ __forceinline__ v16bf frag_a_f32(const float* base, int ld, int row0, int lane) {
    int r = lane & 15, hf = lane >> 4;
    const float* p = base + (size_t)(row0 + r) * ld + hf * 8;
    v4f a0 = *(const v4f*)p, a1 = *(const v4f*)(p + 4);
    v4f b0 = *(const v4f*)(p + 16), b1 = *(const v4f*)(p + 20);
    v16bf f;
#pragma unroll
    for (int i = 0; i < 4; ++i) {
        f[i]      = (bf16_t)a0[i];
        f[i + 4]  = (bf16_t)a1[i];
        f[i + 8]  = (bf16_t)b0[i];
        f[i + 12] = (bf16_t)b1[i];
    }
    return f;
}

// ---------------- fp32 -> bf16 convert -------------------------------------

__global__ void __launch_bounds__(256)
f2b_kernel(const float* __restrict__ x, bf16_t* __restrict__ y, int n) {
    int i = blockIdx.x * 256 + threadIdx.x;
    if (i < n) y[i] = (bf16_t)x[i];
}

// ---------------- generic 128x128 LDS-staged GEMM:  C = A @ Brm^T + bias ----
// A: [M,K] bf16 row-major, Brm: [N,K] bf16 row-major (torch Linear weight).
// MODE 0: store bf16 row-major   MODE 1: store bf16 V-transposed [n,dh,q]
// MODE 2: store fp32 + residual (pre-layernorm)

constexpr int LDS_STRIDE = 40;   // bf16 elems; 80B row pitch avoids bank conflicts

template <int MODE>
__global__ void __launch_bounds__(256)
gemm128_kernel(const bf16_t* __restrict__ A, const bf16_t* __restrict__ Brm,
               const float* __restrict__ bias, bf16_t* __restrict__ Cb,
               float* __restrict__ Cf, const float* __restrict__ resid,
               int K, int N) {
    __shared__ bf16_t Als[128 * LDS_STRIDE];
    __shared__ bf16_t Bls[128 * LDS_STRIDE];

    const int tid = threadIdx.x;
    const int lane = tid & 31, wave = tid >> 5;
    const int wm = wave >> 2, wn = wave & 3;        // 2x4 waves -> wave tile 64x32
    const int rowBase = blockIdx.y * 128;
    const int colBase = blockIdx.x * 128;

    v8f acc[4][2] = {};

    // staging: each thread owns half a row (16 bf16 = 32B) of both tiles
    const int si = tid >> 1;
    const int sh = (tid & 1) * 16;
    const bf16_t* gA = A + (size_t)(rowBase + si) * K + sh;
    const bf16_t* gB = Brm + (size_t)(colBase + si) * K + sh;
    bf16_t* sA = &Als[si * LDS_STRIDE + sh];
    bf16_t* sB = &Bls[si * LDS_STRIDE + sh];

    for (int kk = 0; kk < K; kk += 32) {
        if (kk + 32 < K) {  // pull next tile toward the caches (global_prefetch_b8)
            __builtin_prefetch(gA + kk + 32, 0, 1);
            __builtin_prefetch(gB + kk + 32, 0, 1);
        }
        __syncthreads();                 // previous iteration's readers done
        // direct global -> LDS copies, tracked by ASYNCcnt
        async_copy16B(gA + kk,      sA);
        async_copy16B(gA + kk + 8,  sA + 8);
        async_copy16B(gB + kk,      sB);
        async_copy16B(gB + kk + 8,  sB + 8);
        wait_async0();
        __syncthreads();

        v16bf af[4], bfm[2];
#pragma unroll
        for (int mt = 0; mt < 4; ++mt)
            af[mt] = frag_a(Als, LDS_STRIDE, wm * 64 + mt * 16, lane);
#pragma unroll
        for (int nt = 0; nt < 2; ++nt)
            bfm[nt] = frag_b(Bls, LDS_STRIDE, wn * 32 + nt * 16, lane);
#pragma unroll
        for (int mt = 0; mt < 4; ++mt)
#pragma unroll
            for (int nt = 0; nt < 2; ++nt)
                acc[mt][nt] = wmma_bf16(af[mt], bfm[nt], acc[mt][nt]);
    }

    const int hf = lane >> 4, nn = lane & 15;
#pragma unroll
    for (int mt = 0; mt < 4; ++mt)
#pragma unroll
        for (int nt = 0; nt < 2; ++nt)
#pragma unroll
            for (int e = 0; e < 8; ++e) {
                int row = rowBase + wm * 64 + mt * 16 + e + hf * 8;
                int col = colBase + wn * 32 + nt * 16 + nn;
                float v = acc[mt][nt][e] + bias[col];
                if (MODE == 0) {
                    Cb[(size_t)row * N + col] = (bf16_t)v;
                } else if (MODE == 1) {
                    // flat [4096,1024] -> (n, q, dh); store V transposed: [n, dh, q]
                    int ni = row >> 7;
                    int q  = (row & 127) * 16 + (col >> 6);
                    int dh = col & 63;
                    Cb[(size_t)ni * (DH * Lseq) + (size_t)dh * Lseq + q] = (bf16_t)v;
                } else {
                    size_t o = (size_t)row * N + col;
                    Cf[o] = v + resid[o];
                }
            }
}

// ---------------- scores: S[n] = Q[n] @ K[n]^T * scale  (K-dim = 64) --------

__global__ void __launch_bounds__(256)
scores_kernel(const bf16_t* __restrict__ Qb, const bf16_t* __restrict__ Kb,
              float* __restrict__ attn) {
    const int n = blockIdx.z;
    const int rowBase = blockIdx.y * 128, colBase = blockIdx.x * 128;
    const int tid = threadIdx.x, lane = tid & 31, wave = tid >> 5;
    const int wm = wave >> 2, wn = wave & 3;

    const bf16_t* Qn = Qb + (size_t)n * Lseq * DH;
    const bf16_t* Kn = Kb + (size_t)n * Lseq * DH;

    v8f acc[4][2] = {};
#pragma unroll
    for (int kk = 0; kk < DH; kk += 32) {
        v16bf af[4], bfm[2];
#pragma unroll
        for (int mt = 0; mt < 4; ++mt)
            af[mt] = frag_a(Qn + kk, DH, rowBase + wm * 64 + mt * 16, lane);
#pragma unroll
        for (int nt = 0; nt < 2; ++nt)
            bfm[nt] = frag_b(Kn + kk, DH, colBase + wn * 32 + nt * 16, lane);
#pragma unroll
        for (int mt = 0; mt < 4; ++mt)
#pragma unroll
            for (int nt = 0; nt < 2; ++nt)
                acc[mt][nt] = wmma_bf16(af[mt], bfm[nt], acc[mt][nt]);
    }

    const float scale = 0.125f;   // 64^-0.5
    const int hf = lane >> 4, nn = lane & 15;
    float* An = attn + (size_t)n * Lseq * Lseq;
#pragma unroll
    for (int mt = 0; mt < 4; ++mt)
#pragma unroll
        for (int nt = 0; nt < 2; ++nt)
#pragma unroll
            for (int e = 0; e < 8; ++e) {
                int row = rowBase + wm * 64 + mt * 16 + e + hf * 8;
                int col = colBase + wn * 32 + nt * 16 + nn;
                An[(size_t)row * Lseq + col] = acc[mt][nt][e] * scale;
            }
}

// ---------------- softmax over rows of 2048 (in place in d_out) -------------

__global__ void __launch_bounds__(256)
softmax_kernel(float* __restrict__ attn) {
    __shared__ float red[8];
    const size_t base = (size_t)blockIdx.x * Lseq;
    const int tid = threadIdx.x;

    float v[8];
    float m = -3.4e38f;
#pragma unroll
    for (int j = 0; j < 8; ++j) { v[j] = attn[base + tid + j * 256]; m = fmaxf(m, v[j]); }
#pragma unroll
    for (int off = 16; off > 0; off >>= 1) m = fmaxf(m, __shfl_xor(m, off, 32));
    if ((tid & 31) == 0) red[tid >> 5] = m;
    __syncthreads();
    float bm = red[0];
#pragma unroll
    for (int w = 1; w < 8; ++w) bm = fmaxf(bm, red[w]);
    __syncthreads();

    float s = 0.f;
#pragma unroll
    for (int j = 0; j < 8; ++j) { v[j] = __expf(v[j] - bm); s += v[j]; }
#pragma unroll
    for (int off = 16; off > 0; off >>= 1) s += __shfl_xor(s, off, 32);
    if ((tid & 31) == 0) red[tid >> 5] = s;
    __syncthreads();
    float bs = 0.f;
#pragma unroll
    for (int w = 0; w < 8; ++w) bs += red[w];
    float inv = 1.f / bs;
#pragma unroll
    for (int j = 0; j < 8; ++j) attn[base + tid + j * 256] = v[j] * inv;
}

// ---------------- context: C[n] = P[n] @ V[n]  (P fp32, Vt bf16 [n,dh,q]) ---

__global__ void __launch_bounds__(256)
pv_kernel(const float* __restrict__ attn, const bf16_t* __restrict__ Vt,
          bf16_t* __restrict__ Ctx) {
    const int n = blockIdx.z;
    const int rowBase = blockIdx.x * 128;
    const int tid = threadIdx.x, lane = tid & 31, wave = tid >> 5;
    const int wm = wave >> 1, wn = wave & 1;   // 4x2 waves -> wave tile 32x32

    const float*  Pn = attn + (size_t)n * Lseq * Lseq;
    const bf16_t* Vn = Vt + (size_t)n * DH * Lseq;

    v8f acc[2][2] = {};
    for (int kk = 0; kk < Lseq; kk += 32) {
        v16bf af[2], bfm[2];
#pragma unroll
        for (int mt = 0; mt < 2; ++mt)
            af[mt] = frag_a_f32(Pn + kk, Lseq, rowBase + wm * 32 + mt * 16, lane);
#pragma unroll
        for (int nt = 0; nt < 2; ++nt)
            bfm[nt] = frag_b(Vn + kk, Lseq, wn * 32 + nt * 16, lane);
#pragma unroll
        for (int mt = 0; mt < 2; ++mt)
#pragma unroll
            for (int nt = 0; nt < 2; ++nt)
                acc[mt][nt] = wmma_bf16(af[mt], bfm[nt], acc[mt][nt]);
    }

    const int hf = lane >> 4, nn = lane & 15;
#pragma unroll
    for (int mt = 0; mt < 2; ++mt)
#pragma unroll
        for (int nt = 0; nt < 2; ++nt)
#pragma unroll
            for (int e = 0; e < 8; ++e) {
                int row = rowBase + wm * 32 + mt * 16 + e + hf * 8;
                int col = wn * 32 + nt * 16 + nn;
                Ctx[((size_t)n * Lseq + row) * DH + col] = (bf16_t)acc[mt][nt][e];
            }
}

// ---------------- layernorm over D=1024 (input: pre-LN fp32 w/ residual) ----

__global__ void __launch_bounds__(256)
ln_kernel(const float* __restrict__ pre, const float* __restrict__ gamma,
          const float* __restrict__ beta, float* __restrict__ out) {
    __shared__ float rs[8], rs2[8];
    const size_t base = (size_t)blockIdx.x * Dmod;
    const int tid = threadIdx.x;

    float v[4], s = 0.f, s2 = 0.f;
#pragma unroll
    for (int j = 0; j < 4; ++j) {
        v[j] = pre[base + tid + j * 256];
        s += v[j]; s2 += v[j] * v[j];
    }
#pragma unroll
    for (int off = 16; off > 0; off >>= 1) {
        s  += __shfl_xor(s, off, 32);
        s2 += __shfl_xor(s2, off, 32);
    }
    if ((tid & 31) == 0) { rs[tid >> 5] = s; rs2[tid >> 5] = s2; }
    __syncthreads();
    float S = 0.f, S2 = 0.f;
#pragma unroll
    for (int w = 0; w < 8; ++w) { S += rs[w]; S2 += rs2[w]; }
    float mu  = S * (1.f / Dmod);
    float var = S2 * (1.f / Dmod) - mu * mu;
    float r = rsqrtf(var + 1e-5f);
#pragma unroll
    for (int j = 0; j < 4; ++j) {
        int c = tid + j * 256;
        out[base + c] = (v[j] - mu) * r * gamma[c] + beta[c];
    }
}

// ---------------- host-side orchestration ----------------------------------

extern "C" void kernel_launch(void* const* d_in, const int* in_sizes, int n_in,
                              void* d_out, int out_size, void* d_ws, size_t ws_size,
                              hipStream_t stream) {
    const float* query = (const float*)d_in[0];
    const float* key_  = (const float*)d_in[1];
    const float* value = (const float*)d_in[2];
    const float* Wq = (const float*)d_in[3];
    const float* bq = (const float*)d_in[4];
    const float* Wk = (const float*)d_in[5];
    const float* bk = (const float*)d_in[6];
    const float* Wv = (const float*)d_in[7];
    const float* bv = (const float*)d_in[8];
    const float* Wo = (const float*)d_in[9];
    const float* bo = (const float*)d_in[10];
    const float* gamma = (const float*)d_in[11];
    const float* beta  = (const float*)d_in[12];

    float* outp = (float*)d_out;
    float* attn = (float*)d_out + X_ELEMS;   // [32, 2048, 2048] fp32

    // workspace layout (bytes)
    char* ws = (char*)d_ws;
    const size_t XB = X_ELEMS * sizeof(bf16_t);     // 8 MB
    const size_t WB = W_ELEMS * sizeof(bf16_t);     // 2 MB
    bf16_t* Xq  = (bf16_t*)(ws);
    bf16_t* Xk  = (bf16_t*)(ws + XB);
    bf16_t* Xv  = (bf16_t*)(ws + 2 * XB);
    bf16_t* Wqb = (bf16_t*)(ws + 3 * XB);
    bf16_t* Wkb = (bf16_t*)(ws + 3 * XB + WB);
    bf16_t* Wvb = (bf16_t*)(ws + 3 * XB + 2 * WB);
    bf16_t* Wob = (bf16_t*)(ws + 3 * XB + 3 * WB);
    bf16_t* Qb  = (bf16_t*)(ws + 3 * XB + 4 * WB);
    bf16_t* Kb  = (bf16_t*)(ws + 4 * XB + 4 * WB);
    bf16_t* Vt  = (bf16_t*)(ws + 5 * XB + 4 * WB);  // [n, dh, q]
    bf16_t* Ctx = (bf16_t*)(ws + 6 * XB + 4 * WB);
    float*  Pre = (float*) (ws + 7 * XB + 4 * WB);  // 16 MB fp32

    // 1) fp32 -> bf16 conversions
    f2b_kernel<<<(int)((X_ELEMS + 255) / 256), 256, 0, stream>>>(query, Xq, (int)X_ELEMS);
    f2b_kernel<<<(int)((X_ELEMS + 255) / 256), 256, 0, stream>>>(key_,  Xk, (int)X_ELEMS);
    f2b_kernel<<<(int)((X_ELEMS + 255) / 256), 256, 0, stream>>>(value, Xv, (int)X_ELEMS);
    f2b_kernel<<<(int)((W_ELEMS + 255) / 256), 256, 0, stream>>>(Wq, Wqb, (int)W_ELEMS);
    f2b_kernel<<<(int)((W_ELEMS + 255) / 256), 256, 0, stream>>>(Wk, Wkb, (int)W_ELEMS);
    f2b_kernel<<<(int)((W_ELEMS + 255) / 256), 256, 0, stream>>>(Wv, Wvb, (int)W_ELEMS);
    f2b_kernel<<<(int)((W_ELEMS + 255) / 256), 256, 0, stream>>>(Wo, Wob, (int)W_ELEMS);

    // 2) Q/K/V projections: [4096,1024] @ [1024,1024]^T
    dim3 gp(Dmod / 128, Mrows / 128, 1);
    gemm128_kernel<0><<<gp, 256, 0, stream>>>(Xq, Wqb, bq, Qb, nullptr, nullptr, Dmod, Dmod);
    gemm128_kernel<0><<<gp, 256, 0, stream>>>(Xk, Wkb, bk, Kb, nullptr, nullptr, Dmod, Dmod);
    gemm128_kernel<1><<<gp, 256, 0, stream>>>(Xv, Wvb, bv, Vt, nullptr, nullptr, Dmod, Dmod);

    // 3) scores + softmax
    dim3 gs(Lseq / 128, Lseq / 128, 32);
    scores_kernel<<<gs, 256, 0, stream>>>(Qb, Kb, attn);
    softmax_kernel<<<32 * Lseq, 256, 0, stream>>>(attn);

    // 4) context = P @ V
    dim3 gc(Lseq / 128, 1, 32);
    pv_kernel<<<gc, 256, 0, stream>>>(attn, Vt, Ctx);

    // 5) output projection + residual, then layernorm
    gemm128_kernel<2><<<gp, 256, 0, stream>>>(Ctx, Wob, bo, nullptr, Pre, query, Dmod, Dmod);
    ln_kernel<<<Mrows, 256, 0, stream>>>(Pre, gamma, beta, outp);
}
